// BIISI_42966852829369
// MI455X (gfx1250) — matrix operand order repaired and checked by reading, not compiled, not run on hardware
//
#include <hip/hip_runtime.h>

#define NN 384
#define BATCH 2
#define EE 64
#define EPSF 1e-8f

typedef __attribute__((ext_vector_type(16))) _Float16 v16h;
typedef __attribute__((ext_vector_type(8)))  float    v8f;

// ---------------- WMMA fragment gathers (CDNA5 wave32 layouts) ----------------
// A (16x32 f16): lane L: M=L%16; halves 0..7 -> K = 8*(L/16)+j ; halves 8..15 -> K = 16+8*(L/16)+(j-8)
__device__ __forceinline__ v16h load_a_frag(const _Float16* W, int kStride, int rowBase, int kBase) {
  int lane = threadIdx.x & 31;
  int g = lane >> 4;
  const _Float16* row = W + (rowBase + (lane & 15)) * kStride + kBase;
  v16h a;
#pragma unroll
  for (int j = 0; j < 8; ++j) a[j] = row[8 * g + j];
#pragma unroll
  for (int j = 0; j < 8; ++j) a[8 + j] = row[16 + 8 * g + j];
  return a;
}
// B (32x16 f16): lane L: Ncol=L%16; halves 0..7 -> K = 8*(L/16)+j ; halves 8..15 -> K = 16+8*(L/16)+(j-8) ; X stored [16 spatial][kStride]
__device__ __forceinline__ v16h load_b_frag(const _Float16* X, int kStride, int kBase) {
  int lane = threadIdx.x & 31;
  int g = lane >> 4;
  const _Float16* row = X + (lane & 15) * kStride + kBase;
  v16h b;
#pragma unroll
  for (int j = 0; j < 8; ++j) b[j] = row[8 * g + j];
#pragma unroll
  for (int j = 0; j < 8; ++j) b[8 + j] = row[16 + 8 * g + j];
  return b;
}
#define WMMA_F16(A, B, C) __builtin_amdgcn_wmma_f32_16x16x32_f16(false, (A), false, (B), (short)0, (C), false, false)

// ================= Kernel A: fused per-pair MLP stack =================
// grid = BATCH*96, block = 128 (4 waves). wave handles one src row m, all 384 tgt cols.
__global__ __launch_bounds__(128) void pair_kernel(
    const float* __restrict__ src_emb, const float* __restrict__ tgt_emb,
    const float* __restrict__ tgt, const float* __restrict__ src_cur,
    const float* __restrict__ td_w1, const float* __restrict__ td_b1,
    const float* __restrict__ td_w2, const float* __restrict__ td_b2,
    const float* __restrict__ hd_w1, const float* __restrict__ hd_b1,
    const float* __restrict__ hd_w2, const float* __restrict__ hd_b2,
    const float* __restrict__ sm_w1, const float* __restrict__ sm_b1,
    const float* __restrict__ sm_w2, const float* __restrict__ sm_b2,
    const float* __restrict__ sm2_w1, const float* __restrict__ sm2_b1,
    const float* __restrict__ sm2_w2, const float* __restrict__ sm2_b2,
    float* __restrict__ w_pre, int* __restrict__ corr_idx, int iter)
{
  __shared__ _Float16 sW_hd1[32 * 224];
  __shared__ _Float16 sW_hd2[32 * 32];
  __shared__ _Float16 sW_td1[32 * 32];
  __shared__ _Float16 sW_td2[32 * 32];
  __shared__ _Float16 sW_sm1[32 * 32];
  __shared__ _Float16 sW_sm2[16 * 32];
  __shared__ float    sB[176];        // hd_b1(0) hd_b2(32) td_b1(64) td_b2(96) sm_b1(128) sm_b2(160..175)
  __shared__ float    sS2w1[256];
  __shared__ float    sS2[33];        // sm2_b1(0..15) sm2_w2(16..31) sm2_b2(32)
  __shared__ _Float16 bufA[4][16 * 224];   // per-wave X_hd
  __shared__ _Float16 bufB[4][16 * 32];    // per-wave X_td / prod / sim
  __shared__ _Float16 bufC[4][16 * 32];    // per-wave hidden

  int tid = threadIdx.x;
  // ---- stage weights (f16, zero-padded K) ----
  for (int idx = tid; idx < 32 * 224; idx += 128) {
    int o = idx / 224, k = idx % 224;
    sW_hd1[idx] = (k < 193) ? (_Float16)hd_w1[iter * 32 * 193 + o * 193 + k] : (_Float16)0.f;
  }
  for (int idx = tid; idx < 32 * 32; idx += 128) {
    int o = idx >> 5, k = idx & 31;
    sW_hd2[idx] = (_Float16)hd_w2[iter * 1024 + idx];
    sW_td1[idx] = (k < 10) ? (_Float16)td_w1[iter * 320 + o * 10 + k] : (_Float16)0.f;
    sW_td2[idx] = (_Float16)td_w2[iter * 1024 + idx];
    sW_sm1[idx] = (_Float16)sm_w1[iter * 1024 + idx];
  }
  for (int idx = tid; idx < 16 * 32; idx += 128) sW_sm2[idx] = (_Float16)sm_w2[iter * 512 + idx];
  if (tid < 32) {
    sB[tid]       = hd_b1[iter * 32 + tid];
    sB[32 + tid]  = hd_b2[iter * 32 + tid];
    sB[64 + tid]  = td_b1[iter * 32 + tid];
    sB[96 + tid]  = td_b2[iter * 32 + tid];
    sB[128 + tid] = sm_b1[iter * 32 + tid];
  }
  if (tid < 16) {
    sB[160 + tid] = sm_b2[iter * 16 + tid];
    sS2[tid]      = sm2_b1[iter * 16 + tid];
    sS2[16 + tid] = sm2_w2[iter * 16 + tid];
  }
  for (int idx = tid; idx < 256; idx += 128) sS2w1[idx] = sm2_w1[iter * 256 + idx];
  if (tid == 0) sS2[32] = sm2_b2[iter];
  __syncthreads();

  int b    = blockIdx.x / 96;
  int m    = (blockIdx.x % 96) * 4 + (tid >> 5);
  int lane = tid & 31;
  int g    = lane >> 4;
  int s    = lane & 15;
  int wave = tid >> 5;
  _Float16* Xhd  = bufA[wave];
  _Float16* Bsm  = bufB[wave];
  _Float16* Hbuf = bufC[wave];

  const float* semb = src_emb + b * EE * NN;
  const float* temb = tgt_emb + b * EE * NN;
  const float* scx  = src_cur + b * 3 * NN;
  const float* tgx  = tgt     + b * 3 * NN;

  v8f maxc;
#pragma unroll
  for (int r = 0; r < 8; ++r) maxc[r] = -3.4e38f;
  float best = -3.4e38f;
  int bestn = 0;

  for (int ct = 0; ct < 24; ++ct) {
    int n0 = ct * 16;
    if (ct + 1 < 24) __builtin_prefetch(temb + (n0 + 16), 0, 3);
    // -------- build feature panels --------
    if (lane < 16) {           // X_hd (224 x 16) by lanes 0..15
      int n = n0 + lane;
      _Float16* row = Xhd + lane * 224;
      float d2 = 0.f;
      for (int c = 0; c < EE; ++c) {
        float d = semb[c * NN + m] - temb[c * NN + n];
        d2 += d * d;
      }
      float dist = sqrtf(d2);
      float inv = 1.f / (dist + EPSF);
      for (int c = 0; c < EE; ++c) {
        float se = semb[c * NN + m];
        float te = temb[c * NN + n];
        row[c]        = (_Float16)se;
        row[64 + c]   = (_Float16)te;
        row[128 + c]  = (_Float16)((se - te) * inv);
      }
      row[192] = (_Float16)dist;
#pragma unroll
      for (int c = 193; c < 224; ++c) row[c] = (_Float16)0.f;
    } else {                   // X_td (32 x 16) by lanes 16..31, staged in Bsm
      int sl = lane - 16;
      int n = n0 + sl;
      _Float16* row = Bsm + sl * 32;
      float sx = scx[0 * NN + m], sy = scx[1 * NN + m], sz = scx[2 * NN + m];
      float tx = tgx[0 * NN + n], ty = tgx[1 * NN + n], tz = tgx[2 * NN + n];
      float dx = sx - tx, dy = sy - ty, dz = sz - tz;
      float dist = sqrtf(dx * dx + dy * dy + dz * dz);
      float inv = 1.f / (dist + EPSF);
      row[0] = (_Float16)sx; row[1] = (_Float16)sy; row[2] = (_Float16)sz;
      row[3] = (_Float16)tx; row[4] = (_Float16)ty; row[5] = (_Float16)tz;
      row[6] = (_Float16)(dx * inv); row[7] = (_Float16)(dy * inv); row[8] = (_Float16)(dz * inv);
      row[9] = (_Float16)dist;
#pragma unroll
      for (int c = 10; c < 32; ++c) row[c] = (_Float16)0.f;
    }

    // -------- hd layer1: 224 -> 32 (relu) --------
#pragma unroll
    for (int ot = 0; ot < 2; ++ot) {
      v8f acc = {};
      for (int kt = 0; kt < 7; ++kt) {
        v16h a  = load_a_frag(sW_hd1, 224, ot * 16, kt * 32);
        v16h bf = load_b_frag(Xhd, 224, kt * 32);
        acc = WMMA_F16(a, bf, acc);
      }
      _Float16* hrow = Hbuf + s * 32 + ot * 16 + 8 * g;
#pragma unroll
      for (int r = 0; r < 8; ++r)
        hrow[r] = (_Float16)fmaxf(acc[r] + sB[ot * 16 + 8 * g + r], 0.f);
    }
    // -------- hd layer2: 32 -> 32 --------
    v8f chd[2];
#pragma unroll
    for (int ot = 0; ot < 2; ++ot) {
      v8f acc = {};
      v16h a  = load_a_frag(sW_hd2, 32, ot * 16, 0);
      v16h bf = load_b_frag(Hbuf, 32, 0);
      chd[ot] = WMMA_F16(a, bf, acc);
    }
    // -------- td layer1: 32 -> 32 (relu) --------
#pragma unroll
    for (int ot = 0; ot < 2; ++ot) {
      v8f acc = {};
      v16h a  = load_a_frag(sW_td1, 32, ot * 16, 0);
      v16h bf = load_b_frag(Bsm, 32, 0);
      acc = WMMA_F16(a, bf, acc);
      _Float16* hrow = Hbuf + s * 32 + ot * 16 + 8 * g;
#pragma unroll
      for (int r = 0; r < 8; ++r)
        hrow[r] = (_Float16)fmaxf(acc[r] + sB[64 + ot * 16 + 8 * g + r], 0.f);
    }
    // -------- td layer2: 32 -> 32 --------
    v8f ctd[2];
#pragma unroll
    for (int ot = 0; ot < 2; ++ot) {
      v8f acc = {};
      v16h a  = load_a_frag(sW_td2, 32, ot * 16, 0);
      v16h bf = load_b_frag(Hbuf, 32, 0);
      ctd[ot] = WMMA_F16(a, bf, acc);
    }
    // -------- product (sim_hd * sim_td) -> f16 panel --------
#pragma unroll
    for (int ot = 0; ot < 2; ++ot) {
      _Float16* prow = Bsm + s * 32 + ot * 16 + 8 * g;
#pragma unroll
      for (int r = 0; r < 8; ++r) {
        float vh = chd[ot][r] + sB[32 + ot * 16 + 8 * g + r];
        float vt = ctd[ot][r] + sB[96 + ot * 16 + 8 * g + r];
        prow[r] = (_Float16)(vh * vt);
      }
    }
    // -------- sm layer1: 32 -> 32 (relu) --------
#pragma unroll
    for (int ot = 0; ot < 2; ++ot) {
      v8f acc = {};
      v16h a  = load_a_frag(sW_sm1, 32, ot * 16, 0);
      v16h bf = load_b_frag(Bsm, 32, 0);
      acc = WMMA_F16(a, bf, acc);
      _Float16* hrow = Hbuf + s * 32 + ot * 16 + 8 * g;
#pragma unroll
      for (int r = 0; r < 8; ++r)
        hrow[r] = (_Float16)fmaxf(acc[r] + sB[128 + ot * 16 + 8 * g + r], 0.f);
    }
    // -------- sm layer2: 32 -> 16  => sim (16ch x 16sp) --------
    v8f csim;
    {
      v8f acc = {};
      v16h a  = load_a_frag(sW_sm2, 32, 0, 0);
      v16h bf = load_b_frag(Hbuf, 32, 0);
      csim = WMMA_F16(a, bf, acc);
    }
#pragma unroll
    for (int r = 0; r < 8; ++r) {
      float v = csim[r] + sB[160 + 8 * g + r];
      csim[r] = v;
      maxc[r] = fmaxf(maxc[r], v);
    }
    // sim -> f16 panel for scorer
    {
      _Float16* srow = Bsm + s * 16 + 8 * g;
#pragma unroll
      for (int r = 0; r < 8; ++r) srow[r] = (_Float16)csim[r];
    }
    // -------- sm2 scorer (16->16 relu ->1), clip, argmax over n --------
    {
      const _Float16* srow = Bsm + s * 16;
      float val = sS2[32];
      for (int h = 0; h < 16; ++h) {
        float a = sS2[h];
#pragma unroll
        for (int c = 0; c < 16; ++c) a += sS2w1[h * 16 + c] * (float)srow[c];
        val += sS2[16 + h] * fmaxf(a, 0.f);
      }
      val = fminf(fmaxf(val, -20.f), 20.f);
      int n = n0 + s;
      if (val > best || (val == best && n < bestn)) { best = val; bestn = n; }
    }
  } // ct

  // cross-lane reductions within 16-lane halves (wave32)
#pragma unroll
  for (int off = 1; off < 16; off <<= 1) {
#pragma unroll
    for (int r = 0; r < 8; ++r) maxc[r] = fmaxf(maxc[r], __shfl_xor(maxc[r], off, 32));
  }
#pragma unroll
  for (int off = 1; off < 16; off <<= 1) {
    float ob = __shfl_xor(best, off, 32);
    int   on = __shfl_xor(bestn, off, 32);
    if (ob > best || (ob == best && on < bestn)) { best = ob; bestn = on; }
  }
  if (s == 0) {
#pragma unroll
    for (int r = 0; r < 8; ++r) w_pre[(b * 16 + (8 * g + r)) * NN + m] = maxc[r];
    if (g == 0) corr_idx[b * NN + m] = bestn;
  }
}

// ================= Kernel F: frerc via FFT sparsity =================
// fft2(a_m - a_n) is nonzero only on row 0 / column 0 => O(N^2) DFT + O(N) MLP chains.
__device__ void fr_chain(const float* w3s, const float* w4s, const float* w5s, const float* w6s,
                         const float* gr, const float* gi, float* out)
{
  float y[16];
#pragma unroll
  for (int o = 0; o < 16; ++o) y[o] = 0.f;
  float t1[16];
#pragma unroll
  for (int o = 0; o < 16; ++o) {
    float a = w3s[o * 3] * gi[0] + w3s[o * 3 + 1] * gi[1] + w3s[o * 3 + 2] * gi[2];
    t1[o] = (a > 0.f) ? a : 0.2f * a;
  }
#pragma unroll
  for (int o = 0; o < 16; ++o) {
    float a = 0.f;
    for (int c = 0; c < 16; ++c) a += w4s[o * 16 + c] * t1[c];
    y[o] += a;
  }
#pragma unroll
  for (int o = 0; o < 16; ++o) {
    float a = w3s[o * 3] * gr[0] + w3s[o * 3 + 1] * gr[1] + w3s[o * 3 + 2] * gr[2];
    t1[o] = (a > 0.f) ? a : 0.2f * a;
  }
#pragma unroll
  for (int o = 0; o < 16; ++o) {
    float a = 0.f;
    for (int c = 0; c < 16; ++c) a += w4s[o * 16 + c] * t1[c];
    y[o] += a;
  }
  float t2[16];
#pragma unroll
  for (int o = 0; o < 16; ++o) {
    float a = 0.f;
    for (int c = 0; c < 16; ++c) a += w5s[o * 16 + c] * y[c];
    t2[o] = (a > 0.f) ? a : 0.2f * a;
  }
#pragma unroll
  for (int o = 0; o < 16; ++o) {
    float a = 0.f;
    for (int c = 0; c < 16; ++c) a += w6s[o * 16 + c] * t2[c];
    out[o] = a;
  }
}

__global__ __launch_bounds__(384) void frerc_kernel(
    const float* __restrict__ src_cur, const float* __restrict__ tgt,
    const float* __restrict__ fr_w3, const float* __restrict__ fr_w4,
    const float* __restrict__ fr_w5, const float* __restrict__ fr_w6,
    const float* __restrict__ wrc_w1, const float* __restrict__ wrc_b1,
    const float* __restrict__ wrc_w2, const float* __restrict__ wrc_b2,
    float* __restrict__ factor)
{
  __shared__ float w3s[48], w4s[256], w5s[256], w6s[256], rw1s[256], rb1s[16], rw2s[16];
  __shared__ float redbuf[384];
  __shared__ float rowmax[16];
  int b = blockIdx.x, k = threadIdx.x;
  if (k < 48) w3s[k] = fr_w3[k];
  if (k < 256) { w4s[k] = fr_w4[k]; w5s[k] = fr_w5[k]; w6s[k] = fr_w6[k]; rw1s[k] = wrc_w1[k]; }
  if (k < 16) { rb1s[k] = wrc_b1[k]; rw2s[k] = wrc_w2[k]; }
  __syncthreads();

  const float* scur = src_cur + b * 3 * NN;
  const float* tg   = tgt     + b * 3 * NN;
  float Ar[3] = {0, 0, 0}, Ai[3] = {0, 0, 0}, Tr[3] = {0, 0, 0}, Ti[3] = {0, 0, 0};
  for (int mm = 0; mm < NN; ++mm) {
    float ang = -6.283185307179586f * (float)((k * mm) % NN) / (float)NN;
    float sn, cs;
    sincosf(ang, &sn, &cs);
#pragma unroll
    for (int c = 0; c < 3; ++c) {
      float sv = scur[c * NN + mm], tv = tg[c * NN + mm];
      Ar[c] += sv * cs; Ai[c] += sv * sn;
      Tr[c] += tv * cs; Ti[c] += tv * sn;
    }
  }
  float gr[3], gi[3], grn[3], gin[3];
#pragma unroll
  for (int c = 0; c < 3; ++c) {
    gr[c] = (k == 0) ? 0.f : (float)NN * (Ar[c] - Tr[c]);
    gi[c] = (k == 0) ? 0.f : (float)NN * (Ai[c] - Ti[c]);
    grn[c] = -gr[c]; gin[c] = -gi[c];
  }
  float colz[16], rowz[16];
  fr_chain(w3s, w4s, w5s, w6s, gr, gi, colz);
  fr_chain(w3s, w4s, w5s, w6s, grn, gin, rowz);

  for (int o = 0; o < 16; ++o) {
    redbuf[k] = rowz[o];
    __syncthreads();
    if (k < 128) redbuf[k] = fmaxf(redbuf[k], fmaxf(redbuf[k + 128], redbuf[k + 256]));
    __syncthreads();
    for (int off = 64; off > 0; off >>= 1) {
      if (k < off) redbuf[k] = fmaxf(redbuf[k], redbuf[k + off]);
      __syncthreads();
    }
    if (k == 0) rowmax[o] = redbuf[0];
    __syncthreads();
  }

  float pre[16];
#pragma unroll
  for (int o = 0; o < 16; ++o) pre[o] = (k == 0) ? rowmax[o] : fmaxf(colz[o], 0.f);
  float val = wrc_b2[0];
  for (int h = 0; h < 16; ++h) {
    float a = rb1s[h];
#pragma unroll
    for (int c = 0; c < 16; ++c) a += rw1s[h * 16 + c] * pre[c];
    val += rw2s[h] * fmaxf(a, 0.f);
  }
  factor[b * NN + k] = 1.f - tanhf(fabsf(val));
}

// ================= Kernel B: wf MLP + sigmoid + median + normalize =================
__global__ __launch_bounds__(384) void weight_kernel(
    const float* __restrict__ w_pre, const float* __restrict__ factor,
    const float* __restrict__ wf_w1, const float* __restrict__ wf_b1,
    const float* __restrict__ wf_w2, const float* __restrict__ wf_b2,
    float* __restrict__ wout, int iter, int use_factor)
{
  __shared__ float arr[512];
  int b = blockIdx.x, tid = threadIdx.x;
  float v[16];
#pragma unroll
  for (int c = 0; c < 16; ++c) v[c] = w_pre[(b * 16 + c) * NN + tid];
  float val = wf_b2[iter];
  for (int h = 0; h < 16; ++h) {
    float a = wf_b1[iter * 16 + h];
#pragma unroll
    for (int c = 0; c < 16; ++c) a += wf_w1[iter * 256 + h * 16 + c] * v[c];
    val += wf_w2[iter * 16 + h] * fmaxf(a, 0.f);
  }
  if (use_factor) val *= factor[b * NN + tid];
  float sig = 1.f / (1.f + expf(-val));
  arr[tid] = sig;
  if (tid < 128) arr[384 + tid] = 3.4e38f;
  __syncthreads();
  // bitonic sort (ascending), 512 with +inf padding
  for (int kk = 2; kk <= 512; kk <<= 1) {
    for (int j = kk >> 1; j > 0; j >>= 1) {
      for (int idx = tid; idx < 512; idx += 384) {
        int ixj = idx ^ j;
        if (ixj > idx) {
          bool up = ((idx & kk) == 0);
          float a0 = arr[idx], a1 = arr[ixj];
          if (up ? (a0 > a1) : (a0 < a1)) { arr[idx] = a1; arr[ixj] = a0; }
        }
      }
      __syncthreads();
    }
  }
  float med = arr[(NN - 1) / 2];
  __syncthreads();
  float wthr = (sig >= med) ? sig : 0.f;
  arr[tid] = wthr;
  __syncthreads();
  if (tid < 128) arr[tid] += arr[tid + 256];
  __syncthreads();
  for (int off = 128; off > 0; off >>= 1) {
    if (tid < off) arr[tid] += arr[tid + off];
    __syncthreads();
  }
  wout[b * NN + tid] = wthr / (arr[0] + EPSF);
}

// ================= Kernel C: Kabsch (3x3 SVD) + transform update =================
__global__ __launch_bounds__(128) void svd_kernel(
    const float* __restrict__ tgt, const float* __restrict__ wv,
    const int* __restrict__ corr_idx, float* __restrict__ src_cur,
    float* __restrict__ Rtot, float* __restrict__ ttot)
{
  __shared__ float red[128 * 9];
  __shared__ float sm[3], tm[3], Hs[9], Rm[9], tv[3];
  int b = blockIdx.x, tid = threadIdx.x;
  const float* tg = tgt + b * 3 * NN;
  float* scur = src_cur + b * 3 * NN;
  const float* w = wv + b * NN;
  const int* ci = corr_idx + b * NN;

  // pass 1: weighted centroids
  float acc[6] = {0, 0, 0, 0, 0, 0};
  for (int n = tid; n < NN; n += 128) {
    float wn = w[n];
    int idx = ci[n];
    acc[0] += wn * scur[0 * NN + n]; acc[1] += wn * scur[1 * NN + n]; acc[2] += wn * scur[2 * NN + n];
    acc[3] += wn * tg[0 * NN + idx]; acc[4] += wn * tg[1 * NN + idx]; acc[5] += wn * tg[2 * NN + idx];
  }
  for (int c = 0; c < 6; ++c) red[c * 128 + tid] = acc[c];
  __syncthreads();
  for (int off = 64; off > 0; off >>= 1) {
    if (tid < off) for (int c = 0; c < 6; ++c) red[c * 128 + tid] += red[c * 128 + tid + off];
    __syncthreads();
  }
  if (tid == 0) { for (int c = 0; c < 3; ++c) { sm[c] = red[c * 128]; tm[c] = red[(3 + c) * 128]; } }
  __syncthreads();

  // pass 2: weighted covariance H
  float h[9] = {0, 0, 0, 0, 0, 0, 0, 0, 0};
  for (int n = tid; n < NN; n += 128) {
    float wn = w[n];
    int idx = ci[n];
    float sx = scur[0 * NN + n] - sm[0], sy = scur[1 * NN + n] - sm[1], sz = scur[2 * NN + n] - sm[2];
    float tx = tg[0 * NN + idx] - tm[0], ty = tg[1 * NN + idx] - tm[1], tz = tg[2 * NN + idx] - tm[2];
    h[0] += wn * sx * tx; h[1] += wn * sx * ty; h[2] += wn * sx * tz;
    h[3] += wn * sy * tx; h[4] += wn * sy * ty; h[5] += wn * sy * tz;
    h[6] += wn * sz * tx; h[7] += wn * sz * ty; h[8] += wn * sz * tz;
  }
  __syncthreads();
  for (int c = 0; c < 9; ++c) red[c * 128 + tid] = h[c];
  __syncthreads();
  for (int off = 64; off > 0; off >>= 1) {
    if (tid < off) for (int c = 0; c < 9; ++c) red[c * 128 + tid] += red[c * 128 + tid + off];
    __syncthreads();
  }
  if (tid == 0) for (int c = 0; c < 9; ++c) Hs[c] = red[c * 128];
  __syncthreads();

  if (tid == 0) {
    float A0[9];
    for (int i = 0; i < 9; ++i) A0[i] = Hs[i];
    // S = A^T A, Jacobi eigendecomposition
    float S[3][3], V[3][3] = {{1, 0, 0}, {0, 1, 0}, {0, 0, 1}};
    for (int i = 0; i < 3; ++i)
      for (int j = 0; j < 3; ++j)
        S[i][j] = A0[0 * 3 + i] * A0[0 * 3 + j] + A0[1 * 3 + i] * A0[1 * 3 + j] + A0[2 * 3 + i] * A0[2 * 3 + j];
    for (int sweep = 0; sweep < 36; ++sweep) {
      int r3 = sweep % 3;
      int p = (r3 == 2) ? 1 : 0;
      int q = (r3 == 0) ? 1 : 2;
      float apq = S[p][q];
      if (fabsf(apq) < 1e-20f) continue;
      float theta = (S[q][q] - S[p][p]) / (2.f * apq);
      float tj = copysignf(1.f, theta) / (fabsf(theta) + sqrtf(1.f + theta * theta));
      float cj = 1.f / sqrtf(1.f + tj * tj);
      float sj = tj * cj;
      for (int kk = 0; kk < 3; ++kk) {
        float skp = S[kk][p], skq = S[kk][q];
        S[kk][p] = cj * skp - sj * skq; S[kk][q] = sj * skp + cj * skq;
      }
      for (int kk = 0; kk < 3; ++kk) {
        float spk = S[p][kk], sqk = S[q][kk];
        S[p][kk] = cj * spk - sj * sqk; S[q][kk] = sj * spk + cj * sqk;
      }
      for (int kk = 0; kk < 3; ++kk) {
        float vkp = V[kk][p], vkq = V[kk][q];
        V[kk][p] = cj * vkp - sj * vkq; V[kk][q] = sj * vkp + cj * vkq;
      }
    }
    float lam[3] = {S[0][0], S[1][1], S[2][2]};
    int ordv[3] = {0, 1, 2};
    for (int i = 0; i < 2; ++i)
      for (int j = i + 1; j < 3; ++j)
        if (lam[ordv[j]] > lam[ordv[i]]) { int t2 = ordv[i]; ordv[i] = ordv[j]; ordv[j] = t2; }
    float Vs[3][3], sv[3];
    for (int j = 0; j < 3; ++j) {
      int oj = ordv[j];
      sv[j] = sqrtf(fmaxf(lam[oj], 0.f));
      for (int i = 0; i < 3; ++i) Vs[i][j] = V[i][oj];
    }
    float U[3][3];
    for (int j = 0; j < 3; ++j) {
      float inv = 1.f / fmaxf(sv[j], 1e-12f);
      for (int i = 0; i < 3; ++i)
        U[i][j] = (A0[i * 3 + 0] * Vs[0][j] + A0[i * 3 + 1] * Vs[1][j] + A0[i * 3 + 2] * Vs[2][j]) * inv;
    }
    float r[3][3];
    for (int i = 0; i < 3; ++i)
      for (int kk = 0; kk < 3; ++kk)
        r[i][kk] = Vs[i][0] * U[kk][0] + Vs[i][1] * U[kk][1] + Vs[i][2] * U[kk][2];
    float det = r[0][0] * (r[1][1] * r[2][2] - r[1][2] * r[2][1])
              - r[0][1] * (r[1][0] * r[2][2] - r[1][2] * r[2][0])
              + r[0][2] * (r[1][0] * r[2][1] - r[1][1] * r[2][0]);
    if (det < 0.f) {
      for (int i = 0; i < 3; ++i) Vs[i][2] = -Vs[i][2];
      for (int i = 0; i < 3; ++i)
        for (int kk = 0; kk < 3; ++kk)
          r[i][kk] = Vs[i][0] * U[kk][0] + Vs[i][1] * U[kk][1] + Vs[i][2] * U[kk][2];
    }
    float tvec[3];
    for (int i = 0; i < 3; ++i) tvec[i] = tm[i] - (r[i][0] * sm[0] + r[i][1] * sm[1] + r[i][2] * sm[2]);
    // accumulate totals
    float Rold[9], told[3];
    for (int i = 0; i < 9; ++i) Rold[i] = Rtot[b * 9 + i];
    for (int i = 0; i < 3; ++i) told[i] = ttot[b * 3 + i];
    for (int i = 0; i < 3; ++i)
      for (int kk = 0; kk < 3; ++kk)
        Rtot[b * 9 + i * 3 + kk] = r[i][0] * Rold[0 * 3 + kk] + r[i][1] * Rold[1 * 3 + kk] + r[i][2] * Rold[2 * 3 + kk];
    for (int i = 0; i < 3; ++i)
      ttot[b * 3 + i] = r[i][0] * told[0] + r[i][1] * told[1] + r[i][2] * told[2] + tvec[i];
    for (int i = 0; i < 9; ++i) Rm[i] = r[i / 3][i % 3];
    for (int i = 0; i < 3; ++i) tv[i] = tvec[i];
  }
  __syncthreads();
  // apply transform to src_cur
  for (int n = tid; n < NN; n += 128) {
    float x = scur[0 * NN + n], y = scur[1 * NN + n], z = scur[2 * NN + n];
    scur[0 * NN + n] = Rm[0] * x + Rm[1] * y + Rm[2] * z + tv[0];
    scur[1 * NN + n] = Rm[3] * x + Rm[4] * y + Rm[5] * z + tv[1];
    scur[2 * NN + n] = Rm[6] * x + Rm[7] * y + Rm[8] * z + tv[2];
  }
}

// ================= init =================
__global__ void init_kernel(const float* __restrict__ src, float* __restrict__ src_cur,
                            float* __restrict__ out)
{
  int tid = blockIdx.x * blockDim.x + threadIdx.x;
  if (tid < BATCH * 3 * NN) src_cur[tid] = src[tid];
  if (tid < 18) { int r = tid % 9; out[tid] = (r == 0 || r == 4 || r == 8) ? 1.f : 0.f; }
  else if (tid < 24) out[tid] = 0.f;
}

extern "C" void kernel_launch(void* const* d_in, const int* in_sizes, int n_in,
                              void* d_out, int out_size, void* d_ws, size_t ws_size,
                              hipStream_t stream) {
  (void)in_sizes; (void)n_in; (void)out_size; (void)ws_size;
  const float* src     = (const float*)d_in[0];
  const float* tgt     = (const float*)d_in[1];
  const float* semb    = (const float*)d_in[2];
  const float* temb    = (const float*)d_in[3];
  const float* td_w1   = (const float*)d_in[8];
  const float* td_b1   = (const float*)d_in[9];
  const float* td_w2   = (const float*)d_in[10];
  const float* td_b2   = (const float*)d_in[11];
  const float* hd_w1   = (const float*)d_in[12];
  const float* hd_b1   = (const float*)d_in[13];
  const float* hd_w2   = (const float*)d_in[14];
  const float* hd_b2   = (const float*)d_in[15];
  const float* sm_w1   = (const float*)d_in[16];
  const float* sm_b1   = (const float*)d_in[17];
  const float* sm_w2   = (const float*)d_in[18];
  const float* sm_b2   = (const float*)d_in[19];
  const float* sm2_w1  = (const float*)d_in[20];
  const float* sm2_b1  = (const float*)d_in[21];
  const float* sm2_w2  = (const float*)d_in[22];
  const float* sm2_b2  = (const float*)d_in[23];
  const float* wf_w1   = (const float*)d_in[24];
  const float* wf_b1   = (const float*)d_in[25];
  const float* wf_w2   = (const float*)d_in[26];
  const float* wf_b2   = (const float*)d_in[27];
  const float* wrc_w1  = (const float*)d_in[28];
  const float* wrc_b1  = (const float*)d_in[29];
  const float* wrc_w2  = (const float*)d_in[30];
  const float* wrc_b2  = (const float*)d_in[31];
  const float* fr_w3   = (const float*)d_in[32];
  const float* fr_w4   = (const float*)d_in[33];
  const float* fr_w5   = (const float*)d_in[34];
  const float* fr_w6   = (const float*)d_in[35];

  float* ws      = (float*)d_ws;
  float* src_cur = ws;                         // 2304
  float* w_pre   = ws + 2304;                  // 12288
  float* wbuf    = ws + 2304 + 12288;          // 768
  int*   corr    = (int*)(ws + 15360);         // 768
  float* factor  = ws + 16128;                 // 768
  float* Rtot    = (float*)d_out;              // 18
  float* ttot    = Rtot + 18;                  // 6

  init_kernel<<<dim3(9), 256, 0, stream>>>(src, src_cur, (float*)d_out);

  for (int it = 0; it < 3; ++it) {
    pair_kernel<<<dim3(BATCH * 96), 128, 0, stream>>>(
        semb, temb, tgt, src_cur,
        td_w1, td_b1, td_w2, td_b2,
        hd_w1, hd_b1, hd_w2, hd_b2,
        sm_w1, sm_b1, sm_w2, sm_b2,
        sm2_w1, sm2_b1, sm2_w2, sm2_b2,
        w_pre, corr, it);
    if (it == 2) {
      frerc_kernel<<<dim3(BATCH), 384, 0, stream>>>(
          src_cur, tgt, fr_w3, fr_w4, fr_w5, fr_w6,
          wrc_w1, wrc_b1, wrc_w2, wrc_b2, factor);
    }
    weight_kernel<<<dim3(BATCH), 384, 0, stream>>>(
        w_pre, factor, wf_w1, wf_b1, wf_w2, wf_b2, wbuf, it, (it == 2) ? 1 : 0);
    svd_kernel<<<dim3(BATCH), 128, 0, stream>>>(
        tgt, wbuf, corr, src_cur, Rtot, ttot);
  }
}
